// GAT_27212912787985
// MI455X (gfx1250) — matrix operand order repaired and checked by reading, not compiled, not run on hardware
//
#include <hip/hip_runtime.h>
#include <hip/hip_bf16.h>

// ------------------------------------------------------------------
// 2-layer GAT for MI455X (gfx1250, wave32, WMMA).
// GEMMs -> v_wmma_f32_16x16x32_bf16 (bf16 in, fp32 accumulate).
// Edge softmax + aggregation -> streaming fp32 kernels with atomics
// (memory-bound part; ~2.8 GB irregular traffic dominates runtime).
// ------------------------------------------------------------------

#define NEG_SLOPE 0.2f

typedef __attribute__((ext_vector_type(16))) __bf16 v16bf;
typedef __attribute__((ext_vector_type(8)))  float  v8f;

__device__ __forceinline__ unsigned short f2bf(float f) {
  unsigned u = __float_as_uint(f);
  u += 0x7FFFu + ((u >> 16) & 1u);          // round-to-nearest-even
  return (unsigned short)(u >> 16);
}
// monotone float -> uint key (for atomicMax-based segment max)
__device__ __forceinline__ unsigned fkey(float f) {
  unsigned u = __float_as_uint(f);
  return (u & 0x80000000u) ? ~u : (u | 0x80000000u);
}
__device__ __forceinline__ float fromkey(unsigned k) {
  unsigned u = (k & 0x80000000u) ? (k & 0x7FFFFFFFu) : ~k;
  return __uint_as_float(u);
}

// ---------------- conversion kernels ----------------
__global__ void cvt_f32_bf16(const float* __restrict__ in,
                             unsigned short* __restrict__ out, int n) {
  int i = blockIdx.x * blockDim.x + threadIdx.x;
  if (i < n) out[i] = f2bf(in[i]);
}

// W[K][Nout] fp32 -> Wt[Nout][K] bf16 (so WMMA B columns are contiguous)
__global__ void cvt_transpose_bf16(const float* __restrict__ W,
                                   unsigned short* __restrict__ Wt,
                                   int K, int Nout) {
  int i = blockIdx.x * blockDim.x + threadIdx.x;
  if (i < K * Nout) {
    int k = i / Nout, n = i % Nout;
    Wt[(size_t)n * K + k] = f2bf(W[i]);
  }
}

// ---------------- WMMA GEMM:  C[M][Nout] = A[M][K] * Bt[Nout][K]^T ----------------
union ABPack { v16bf v; uint4 u[2]; };

__global__ void wmma_gemm_bf16(const unsigned short* __restrict__ A,   // [M][K] bf16
                               const unsigned short* __restrict__ Bt,  // [Nout][K] bf16
                               float* __restrict__ C,                  // [M][Nout]
                               int K, int Nout) {
  const int lane = threadIdx.x;        // wave32, one wave per block
  const int half = lane >> 4;          // 0 | 1
  const int l16  = lane & 15;
  const unsigned short* arow = A  + (size_t)(blockIdx.x * 16 + l16) * K;
  const unsigned short* bcol = Bt + (size_t)(blockIdx.y * 16 + l16) * K;

  v8f acc = {};
  for (int k0 = 0; k0 < K; k0 += 32) {
    ABPack a, b;
    // A lane layout (ISA 16-bit 16x32): elems 0-7 = K[k0+half*8 ..), elems 8-15 = K[k0+16+half*8 ..)
    a.u[0] = *(const uint4*)(arow + k0 + half * 8);
    a.u[1] = *(const uint4*)(arow + k0 + 16 + half * 8);
    // B lane layout: column = l16, same K order as A: elems 0-7 = K[k0+half*8 ..), elems 8-15 = K[k0+16+half*8 ..)
    b.u[0] = *(const uint4*)(bcol + k0 + half * 8);
    b.u[1] = *(const uint4*)(bcol + k0 + 16 + half * 8);
    acc = __builtin_amdgcn_wmma_f32_16x16x32_bf16(false, a.v, false, b.v,
                                                  (short)0, acc, false, false);
  }
  // C/D layout: VGPR v -> row tileM*16 + v + half*8, col tileN*16 + l16
  float* crow = C + (size_t)(blockIdx.x * 16 + half * 8) * Nout + blockIdx.y * 16 + l16;
#pragma unroll
  for (int v = 0; v < 8; ++v)
    crow[(size_t)v * Nout] = acc[v];
}

// ---------------- attention scalars:  alpha_s/d[n][h] = <h[n,h,:], a_src/dst[h,:]> ----------------
__global__ void alpha_kernel(const float* __restrict__ h,
                             const float* __restrict__ a_src,
                             const float* __restrict__ a_dst,
                             float* __restrict__ alpS, float* __restrict__ alpD,
                             int Nn, int H, int C) {
  int i = blockIdx.x * blockDim.x + threadIdx.x;   // n*H + head
  if (i >= Nn * H) return;
  int n = i / H, hh = i % H;
  const float* hp = h + (size_t)n * H * C + (size_t)hh * C;
  const float* s  = a_src + hh * C;
  const float* d  = a_dst + hh * C;
  float as = 0.f, ad = 0.f;
  for (int c = 0; c < C; ++c) { float v = hp[c]; as += v * s[c]; ad += v * d[c]; }
  alpS[i] = as; alpD[i] = ad;
}

// ---------------- edge pass 1: segment max of leaky-relu logits over dst ----------------
__global__ void edge_max(const long long* __restrict__ ei, int E, int Nn, int H,
                         const float* __restrict__ alpS, const float* __restrict__ alpD,
                         unsigned* __restrict__ emaxu) {
  int i = blockIdx.x * blockDim.x + threadIdx.x;
  if (i >= E + Nn) return;
  int s, d;
  if (i < E) { s = (int)ei[i]; d = (int)ei[E + i]; } else { s = d = i - E; }
  for (int h = 0; h < H; ++h) {
    float e = alpS[s * H + h] + alpD[d * H + h];
    e = (e > 0.f) ? e : NEG_SLOPE * e;
    atomicMax(&emaxu[d * H + h], fkey(e));
  }
}

// ---------------- edge pass 2: w = exp(e - max); denom[dst] += w ----------------
__global__ void edge_exp(const long long* __restrict__ ei, int E, int Nn, int H,
                         const float* __restrict__ alpS, const float* __restrict__ alpD,
                         const unsigned* __restrict__ emaxu,
                         float* __restrict__ wE, float* __restrict__ denom) {
  int i = blockIdx.x * blockDim.x + threadIdx.x;
  if (i >= E + Nn) return;
  int s, d;
  if (i < E) { s = (int)ei[i]; d = (int)ei[E + i]; } else { s = d = i - E; }
  for (int h = 0; h < H; ++h) {
    float e = alpS[s * H + h] + alpD[d * H + h];
    e = (e > 0.f) ? e : NEG_SLOPE * e;
    float w = __expf(e - fromkey(emaxu[d * H + h]));
    wE[(size_t)i * H + h] = w;
    atomicAdd(&denom[d * H + h], w);
  }
}

// ---------------- edge pass 3: agg[dst] += alpha * h[src]  (one block per edge, H*C threads) ----------------
__global__ void edge_agg(const long long* __restrict__ ei, int E, int Nn, int H, int C,
                         const float* __restrict__ h, const float* __restrict__ wE,
                         const float* __restrict__ denom, float* __restrict__ agg) {
  int i = blockIdx.x;                    // edge id
  int t = threadIdx.x;                   // 0 .. H*C-1 (coalesced channel)
  int s, d;
  if (i < E) { s = (int)ei[i]; d = (int)ei[E + i]; } else { s = d = i - E; }
  int hh = t / C;
  float alpha = wE[(size_t)i * H + hh] / (denom[d * H + hh] + 1e-16f);
  atomicAdd(&agg[(size_t)d * H * C + t], alpha * h[(size_t)s * H * C + t]);
}

// ---------------- finalize layer 1: head-mean + bias + relu -> bf16 input of layer 2 ----------------
__global__ void finalize_relu_bf16(const float* __restrict__ agg, const float* __restrict__ bias,
                                   unsigned short* __restrict__ outbf, int Nn, int H, int C) {
  int i = blockIdx.x * blockDim.x + threadIdx.x;
  if (i >= Nn * C) return;
  int n = i / C, c = i % C;
  float acc = 0.f;
  for (int h = 0; h < H; ++h) acc += agg[(size_t)n * H * C + (size_t)h * C + c];
  float v = acc * (1.0f / H) + bias[c];
  outbf[i] = f2bf(v > 0.f ? v : 0.f);
}

// ---------------- finalize layer 2: head-mean + bias -> fp32 output ----------------
__global__ void finalize_out(const float* __restrict__ agg, const float* __restrict__ bias,
                             float* __restrict__ out, int Nn, int H, int C) {
  int i = blockIdx.x * blockDim.x + threadIdx.x;
  if (i >= Nn * C) return;
  int n = i / C, c = i % C;
  float acc = 0.f;
  for (int h = 0; h < H; ++h) acc += agg[(size_t)n * H * C + (size_t)h * C + c];
  out[i] = acc * (1.0f / H) + bias[c];
}

extern "C" void kernel_launch(void* const* d_in, const int* in_sizes, int n_in,
                              void* d_out, int out_size, void* d_ws, size_t ws_size,
                              hipStream_t stream) {
  const float*     x   = (const float*)d_in[0];
  const long long* ei  = (const long long*)d_in[1];   // int64 edge_index [2][E]
  const float*     W1  = (const float*)d_in[2];
  const float*     as1 = (const float*)d_in[3];
  const float*     ad1 = (const float*)d_in[4];
  const float*     b1  = (const float*)d_in[5];
  const float*     W2  = (const float*)d_in[6];
  const float*     as2 = (const float*)d_in[7];
  const float*     ad2 = (const float*)d_in[8];
  const float*     b2  = (const float*)d_in[9];

  const int FIN = 128, H = 4, HID = 64, NCLS = 40;
  const int Nn   = in_sizes[0] / FIN;   // 50000 (divisible by 16)
  const int E    = in_sizes[1] / 2;     // 800000
  const int Etot = E + Nn;              // + self loops

  // -------- workspace arena (reused across both layers) --------
  char* ws = (char*)d_ws;
  size_t off = 0;
  auto alloc = [&](size_t bytes) -> char* {
    char* p = ws + off;
    off += (bytes + 255) & ~(size_t)255;
    return p;
  };
  float*          h_feat = (float*)alloc((size_t)Nn * H * HID * 4);   // [N, H*C] (max 256)
  float*          agg    = (float*)alloc((size_t)Nn * H * HID * 4);
  unsigned short* xbf    = (unsigned short*)alloc((size_t)Nn * FIN * 2);
  unsigned short* wt     = (unsigned short*)alloc((size_t)FIN * H * HID * 2);
  float*          alpS   = (float*)alloc((size_t)Nn * H * 4);
  float*          alpD   = (float*)alloc((size_t)Nn * H * 4);
  unsigned*       emaxu  = (unsigned*)alloc((size_t)Nn * H * 4);
  float*          denom  = (float*)alloc((size_t)Nn * H * 4);
  float*          wE     = (float*)alloc((size_t)Etot * H * 4);
  (void)ws_size; (void)n_in; (void)out_size;

  // ===================== Layer 1 =====================
  {
    const int C = HID, Nout = H * C, K = FIN;
    int t = Nn * K;
    cvt_f32_bf16<<<(t + 255) / 256, 256, 0, stream>>>(x, xbf, t);
    t = K * Nout;
    cvt_transpose_bf16<<<(t + 255) / 256, 256, 0, stream>>>(W1, wt, K, Nout);

    dim3 g(Nn / 16, Nout / 16);
    wmma_gemm_bf16<<<g, 32, 0, stream>>>(xbf, wt, h_feat, K, Nout);

    t = Nn * H;
    alpha_kernel<<<(t + 255) / 256, 256, 0, stream>>>(h_feat, as1, ad1, alpS, alpD, Nn, H, C);

    hipMemsetAsync(emaxu, 0, (size_t)Nn * H * 4, stream);
    hipMemsetAsync(denom, 0, (size_t)Nn * H * 4, stream);
    hipMemsetAsync(agg,   0, (size_t)Nn * Nout * 4, stream);

    edge_max<<<(Etot + 255) / 256, 256, 0, stream>>>(ei, E, Nn, H, alpS, alpD, emaxu);
    edge_exp<<<(Etot + 255) / 256, 256, 0, stream>>>(ei, E, Nn, H, alpS, alpD, emaxu, wE, denom);
    edge_agg<<<Etot, Nout, 0, stream>>>(ei, E, Nn, H, C, h_feat, wE, denom, agg);

    t = Nn * C;
    finalize_relu_bf16<<<(t + 255) / 256, 256, 0, stream>>>(agg, b1, xbf, Nn, H, C);
  }

  // ===================== Layer 2 =====================
  {
    const int C = NCLS, Nout = H * C, K = HID;
    int t = K * Nout;
    cvt_transpose_bf16<<<(t + 255) / 256, 256, 0, stream>>>(W2, wt, K, Nout);

    dim3 g(Nn / 16, Nout / 16);
    wmma_gemm_bf16<<<g, 32, 0, stream>>>(xbf, wt, h_feat, K, Nout);

    t = Nn * H;
    alpha_kernel<<<(t + 255) / 256, 256, 0, stream>>>(h_feat, as2, ad2, alpS, alpD, Nn, H, C);

    hipMemsetAsync(emaxu, 0, (size_t)Nn * H * 4, stream);
    hipMemsetAsync(denom, 0, (size_t)Nn * H * 4, stream);
    hipMemsetAsync(agg,   0, (size_t)Nn * Nout * 4, stream);

    edge_max<<<(Etot + 255) / 256, 256, 0, stream>>>(ei, E, Nn, H, alpS, alpD, emaxu);
    edge_exp<<<(Etot + 255) / 256, 256, 0, stream>>>(ei, E, Nn, H, alpS, alpD, emaxu, wE, denom);
    edge_agg<<<Etot, Nout, 0, stream>>>(ei, E, Nn, H, C, h_feat, wE, denom, agg);

    t = Nn * C;
    finalize_out<<<(t + 255) / 256, 256, 0, stream>>>(agg, b2, (float*)d_out, Nn, H, C);
  }
}
